// GQA_35055523070279
// MI455X (gfx1250) — matrix operand order repaired and checked by reading, not compiled, not run on hardware
//
#include <hip/hip_runtime.h>
#include <hip/hip_bf16.h>

#define S_LEN 2048
#define DMODEL 2048
#define NH 32
#define NKV 8
#define HDIM 64
#define KVDIM (NKV * HDIM)   // 512

typedef __bf16 bf16;
typedef bf16  v16bf __attribute__((ext_vector_type(16)));
typedef float v8f   __attribute__((ext_vector_type(8)));

union Frag { uint4 u[2]; v16bf v; };

__device__ __forceinline__ v8f wmma_bf16(v16bf a, v16bf b, v8f c) {
  // D = A(16x32 bf16) * B(32x16 bf16) + C(16x16 f32)
  return __builtin_amdgcn_wmma_f32_16x16x32_bf16(false, a, false, b, (short)0, c,
                                                 false, false);
}

__device__ __forceinline__ float lane_read(float v, int srcLane) {
  int r = __builtin_amdgcn_ds_bpermute(srcLane << 2, __builtin_bit_cast(int, v));
  return __builtin_bit_cast(float, r);
}

// ---------------------------------------------------------------- converts
__global__ __launch_bounds__(256) void f32_to_bf16(const float* __restrict__ in,
                                                   bf16* __restrict__ out, int n) {
  int i = blockIdx.x * 256 + threadIdx.x;
  if (i < n) out[i] = (bf16)in[i];
}

// V: [s][512] f32  ->  [512][s] bf16 (per-head column-major for B-frags)
__global__ __launch_bounds__(256) void v_transpose(const float* __restrict__ vf,
                                                   bf16* __restrict__ vt) {
  int i = blockIdx.x * 256 + threadIdx.x;
  if (i >= S_LEN * KVDIM) return;
  int s = i >> 9;          // / 512
  int c = i & (KVDIM - 1); // g*64+d
  vt[(long)c * S_LEN + s] = (bf16)vf[i];
}

// ---------------------------------------------------------------- GEMM
// C[M,N] f32 = A[M,K] bf16 (row-major) * B[N,K] bf16 (row-major)^T
// 256 threads = 8 waves arranged 2(m) x 4(n); each wave computes a 64x64 tile
// as 4x4 WMMA accumulators -> 16 WMMAs per 16 b128 loads per 32-deep k-step.
// Block tile: 128 x 256.
__global__ __launch_bounds__(256) void gemm_bf16(const bf16* __restrict__ A,
                                                 const bf16* __restrict__ B,
                                                 float* __restrict__ C,
                                                 int M, int N, int K) {
  int lane = threadIdx.x & 31, wave = threadIdx.x >> 5;
  int ln = lane & 15, hi = lane >> 4;
  int m0 = blockIdx.y * 128 + (wave >> 2) * 64;
  int n0 = blockIdx.x * 256 + (wave & 3) * 64;

  v8f acc[4][4] = {};

  for (int k0 = 0; k0 < K; k0 += 32) {
    Frag a[4], b[4];
#pragma unroll
    for (int t = 0; t < 4; ++t) {
      const bf16* ap = A + (long)(m0 + t * 16 + ln) * K + k0;
      a[t].u[0] = *(const uint4*)(ap + 8 * hi);
      a[t].u[1] = *(const uint4*)(ap + 16 + 8 * hi);
    }
#pragma unroll
    for (int t = 0; t < 4; ++t) {
      const bf16* bp = B + (long)(n0 + t * 16 + ln) * K + k0;
      b[t].u[0] = *(const uint4*)(bp + 8 * hi);
      b[t].u[1] = *(const uint4*)(bp + 16 + 8 * hi);
    }
    if (k0 + 96 < K) {
#pragma unroll
      for (int t = 0; t < 4; ++t) {
        __builtin_prefetch(A + (long)(m0 + t * 16 + ln) * K + k0 + 96, 0, 1);
        __builtin_prefetch(B + (long)(n0 + t * 16 + ln) * K + k0 + 96, 0, 1);
      }
    }
#pragma unroll
    for (int mt = 0; mt < 4; ++mt)
#pragma unroll
      for (int nt = 0; nt < 4; ++nt)
        acc[mt][nt] = wmma_bf16(a[mt].v, b[nt].v, acc[mt][nt]);
  }

#pragma unroll
  for (int mt = 0; mt < 4; ++mt)
#pragma unroll
    for (int nt = 0; nt < 4; ++nt)
#pragma unroll
      for (int r = 0; r < 8; ++r) {
        int row = m0 + mt * 16 + r + 8 * hi;
        int col = n0 + nt * 16 + ln;
        C[(long)row * N + col] = acc[mt][nt][r];
      }
}

// ---------------------------------------------------------------- RoPE+RMSNorm
__global__ __launch_bounds__(128) void rope_rmsnorm(const float* __restrict__ in,
                                                    const float* __restrict__ cosb,
                                                    const float* __restrict__ sinb,
                                                    const float* __restrict__ w,
                                                    bf16* __restrict__ out,
                                                    int nheads) {
  int tid = blockIdx.x * 128 + threadIdx.x;
  if (tid >= S_LEN * nheads) return;
  int s = tid / nheads, h = tid % nheads;
  const float* xp = in + (long)s * (nheads * HDIM) + h * HDIM;
  float y[HDIM];
  float ss = 0.f;
#pragma unroll
  for (int i = 0; i < HDIM / 2; ++i) {
    float xr = xp[2 * i], xi = xp[2 * i + 1];
    float c = cosb[s * (HDIM / 2) + i], sn = sinb[s * (HDIM / 2) + i];
    float yr = xr * c - xi * sn;
    float yi = xr * sn + xi * c;
    y[2 * i] = yr; y[2 * i + 1] = yi;
    ss += yr * yr + yi * yi;
  }
  float inv = rsqrtf(ss * (1.f / HDIM) + 1e-6f);
  bf16* op = out + (long)s * (nheads * HDIM) + h * HDIM;
#pragma unroll
  for (int d = 0; d < HDIM; ++d) op[d] = (bf16)(y[d] * inv * w[d]);
}

// ---------------------------------------------------------------- attention
// One wave owns 16 q rows of one head; online softmax over k-blocks of 32.
// Computes S^T = K*Q^T so softmaxed P converts in-register to the A-frag of
// the P*V WMMA (no cross-lane transpose).
__global__ __launch_bounds__(128) void attn_flash(const bf16* __restrict__ qb,
                                                  const bf16* __restrict__ kb,
                                                  const bf16* __restrict__ vt,
                                                  float* __restrict__ of) {
  const float scale = 0.125f;  // 1/sqrt(64)
  int lane = threadIdx.x & 31, wave = threadIdx.x >> 5;
  int ln = lane & 15, hi = lane >> 4;
  int h = blockIdx.y, g = h >> 2;  // NREP = 4
  int qbase = (blockIdx.x * 4 + wave) * 16;

  // Q as B-frags (n = q row, K-dim = d): two d-halves of HD=64
  Frag bq0, bq1;
  {
    const bf16* qrow = qb + (long)(qbase + ln) * DMODEL + h * HDIM + 8 * hi;
    bq0.u[0] = *(const uint4*)(qrow);      bq0.u[1] = *(const uint4*)(qrow + 16);
    bq1.u[0] = *(const uint4*)(qrow + 32); bq1.u[1] = *(const uint4*)(qrow + 48);
  }

  v8f o[4] = {};
  float m_run = -1e30f, l_run = 0.f;
  int qg = qbase + ln;
  int nkb = (qbase + 15) / 32 + 1;

  for (int kbi = 0; kbi < nkb; ++kbi) {
    int k32 = kbi * 32;
    // K rows as A-frags: [ktile 0/1][d-half 0/1]
    Frag ak[2][2];
#pragma unroll
    for (int t = 0; t < 2; ++t) {
      const bf16* krow = kb + (long)(k32 + t * 16 + ln) * KVDIM + g * HDIM;
      ak[t][0].u[0] = *(const uint4*)(krow + 8 * hi);
      ak[t][0].u[1] = *(const uint4*)(krow + 16 + 8 * hi);
      ak[t][1].u[0] = *(const uint4*)(krow + 32 + 8 * hi);
      ak[t][1].u[1] = *(const uint4*)(krow + 48 + 8 * hi);
    }
    v8f t0 = {}, t1 = {};
    t0 = wmma_bf16(ak[0][0].v, bq0.v, t0);
    t0 = wmma_bf16(ak[0][1].v, bq1.v, t0);
    t1 = wmma_bf16(ak[1][0].v, bq0.v, t1);
    t1 = wmma_bf16(ak[1][1].v, bq1.v, t1);

    // scale + causal mask; per-lane tile max (lane holds q = ln)
    float s0[8], s1[8];
    float mt = -1e30f;
#pragma unroll
    for (int r = 0; r < 8; ++r) {
      int kp = k32 + r + 8 * hi;
      s0[r] = (kp <= qg) ? t0[r] * scale : -1e30f;
      s1[r] = (kp + 16 <= qg) ? t1[r] * scale : -1e30f;
      mt = fmaxf(mt, fmaxf(s0[r], s1[r]));
    }
    mt = fmaxf(mt, lane_read(mt, lane ^ 16));  // combine k halves
    float m_new = fmaxf(m_run, mt);
    float alpha = __expf(m_run - m_new);

    Frag ap;
    float rs = 0.f;
#pragma unroll
    for (int r = 0; r < 8; ++r) {
      float p0 = __expf(s0[r] - m_new);
      float p1 = __expf(s1[r] - m_new);
      rs += p0 + p1;
      ap.v[r] = (bf16)p0;        // k = 8*hi + r       (tile 0)
      ap.v[8 + r] = (bf16)p1;    // k = 16 + 8*hi + r  (tile 1)
    }
    rs += lane_read(rs, lane ^ 16);
    l_run = l_run * alpha + rs;
    m_run = m_new;

    // rescale O: row q = r + 8*hi lives in VGPR r; stats live in lane q
#pragma unroll
    for (int r = 0; r < 8; ++r) {
      float ab = lane_read(alpha, r + 8 * hi);
#pragma unroll
      for (int dt = 0; dt < 4; ++dt) o[dt][r] *= ab;
    }

    // O += P * V   (V^T stored [d][s]: contiguous B-frag loads)
#pragma unroll
    for (int dt = 0; dt < 4; ++dt) {
      Frag bv;
      const bf16* vrow =
          vt + (long)(g * HDIM + dt * 16 + ln) * S_LEN + k32 + 8 * hi;
      bv.u[0] = *(const uint4*)(vrow);
      bv.u[1] = *(const uint4*)(vrow + 16);
      o[dt] = wmma_bf16(ap.v, bv.v, o[dt]);
    }
  }

  // finalize: divide by row sum, store f32
#pragma unroll
  for (int r = 0; r < 8; ++r) {
    float li = lane_read(l_run, r + 8 * hi);
    float inv = 1.f / li;
    int row = qbase + r + 8 * hi;
#pragma unroll
    for (int dt = 0; dt < 4; ++dt)
      of[(long)row * DMODEL + h * HDIM + dt * 16 + ln] = o[dt][r] * inv;
  }
}

// ---------------------------------------------------------------- gating
__global__ __launch_bounds__(256) void gate_fuse(const float* __restrict__ of,
                                                 const float* __restrict__ gf,
                                                 bf16* __restrict__ ob, int n) {
  int i = blockIdx.x * 256 + threadIdx.x;
  if (i < n) ob[i] = (bf16)(of[i] * (1.f / (1.f + __expf(-gf[i]))));
}

// ---------------------------------------------------------------- launch
extern "C" void kernel_launch(void* const* d_in, const int* in_sizes, int n_in,
                              void* d_out, int out_size, void* d_ws, size_t ws_size,
                              hipStream_t stream) {
  const float* x    = (const float*)d_in[0];
  const float* cosb = (const float*)d_in[1];
  const float* sinb = (const float*)d_in[2];
  const float* wq   = (const float*)d_in[3];
  const float* wk   = (const float*)d_in[4];
  const float* wv   = (const float*)d_in[5];
  const float* wo   = (const float*)d_in[6];
  const float* wg   = (const float*)d_in[7];
  const float* qn   = (const float*)d_in[8];
  const float* kn   = (const float*)d_in[9];
  float* out = (float*)d_out;

  char* ws = (char*)d_ws;
  const size_t SZ_BIG_BF = (size_t)S_LEN * DMODEL * 2;   // 8 MiB
  const size_t SZ_KV_BF  = (size_t)S_LEN * KVDIM * 2;    // 2 MiB
  const size_t SZ_BIG_F  = (size_t)S_LEN * DMODEL * 4;   // 16 MiB
  const size_t SZ_KV_F   = (size_t)S_LEN * KVDIM * 4;    // 4 MiB

  size_t off = 0;
  bf16* xb  = (bf16*)(ws + off); off += SZ_BIG_BF;
  bf16* wqb = (bf16*)(ws + off); off += SZ_BIG_BF;
  bf16* wkb = (bf16*)(ws + off); off += SZ_KV_BF;
  bf16* wvb = (bf16*)(ws + off); off += SZ_KV_BF;
  bf16* wgb = (bf16*)(ws + off); off += SZ_BIG_BF;
  bf16* wob = (bf16*)(ws + off); off += SZ_BIG_BF;
  float* qf = (float*)(ws + off); off += SZ_BIG_F;
  float* kf = (float*)(ws + off); off += SZ_KV_F;
  float* vf = (float*)(ws + off); off += SZ_KV_F;
  float* gf = (float*)(ws + off); off += SZ_BIG_F;
  bf16* qbuf = (bf16*)(ws + off); off += SZ_BIG_BF;
  bf16* kbuf = (bf16*)(ws + off); off += SZ_KV_BF;
  bf16* vt   = (bf16*)(ws + off); off += SZ_KV_BF;
  float* of  = (float*)(ws + off); off += SZ_BIG_F;
  bf16* ob   = (bf16*)(ws + off); off += SZ_BIG_BF;
  (void)ws_size; (void)in_sizes; (void)n_in; (void)out_size;

  int nbig = S_LEN * DMODEL, nkv = KVDIM * DMODEL;
  f32_to_bf16<<<(nbig + 255) / 256, 256, 0, stream>>>(x, xb, nbig);
  f32_to_bf16<<<(nbig + 255) / 256, 256, 0, stream>>>(wq, wqb, nbig);
  f32_to_bf16<<<(nkv + 255) / 256, 256, 0, stream>>>(wk, wkb, nkv);
  f32_to_bf16<<<(nkv + 255) / 256, 256, 0, stream>>>(wv, wvb, nkv);
  f32_to_bf16<<<(nbig + 255) / 256, 256, 0, stream>>>(wg, wgb, nbig);
  f32_to_bf16<<<(nbig + 255) / 256, 256, 0, stream>>>(wo, wob, nbig);

  // Block tile 128x256 (8 waves, 64x64 per wave)
  gemm_bf16<<<dim3(DMODEL / 256, S_LEN / 128), 256, 0, stream>>>(xb, wqb, qf,
      S_LEN, DMODEL, DMODEL);
  gemm_bf16<<<dim3(KVDIM / 256, S_LEN / 128), 256, 0, stream>>>(xb, wkb, kf,
      S_LEN, KVDIM, DMODEL);
  gemm_bf16<<<dim3(KVDIM / 256, S_LEN / 128), 256, 0, stream>>>(xb, wvb, vf,
      S_LEN, KVDIM, DMODEL);
  gemm_bf16<<<dim3(DMODEL / 256, S_LEN / 128), 256, 0, stream>>>(xb, wgb, gf,
      S_LEN, DMODEL, DMODEL);

  rope_rmsnorm<<<(S_LEN * NH + 127) / 128, 128, 0, stream>>>(qf, cosb, sinb, qn,
                                                             qbuf, NH);
  rope_rmsnorm<<<(S_LEN * NKV + 127) / 128, 128, 0, stream>>>(kf, cosb, sinb, kn,
                                                              kbuf, NKV);
  v_transpose<<<(S_LEN * KVDIM + 255) / 256, 256, 0, stream>>>(vf, vt);

  attn_flash<<<dim3(S_LEN / 64, NH), 128, 0, stream>>>(qbuf, kbuf, vt, of);

  gate_fuse<<<(nbig + 255) / 256, 256, 0, stream>>>(of, gf, ob, nbig);

  gemm_bf16<<<dim3(DMODEL / 256, S_LEN / 128), 256, 0, stream>>>(ob, wob, out,
      S_LEN, DMODEL, DMODEL);
}
